// MultiLayerGRU_40707700032438
// MI455X (gfx1250) — compile-verified, matrix operand order checked by reading, not run on hardware
//
#include <hip/hip_runtime.h>
#include <hip/hip_bf16.h>

// Problem constants
#define BB   32
#define TT   512
#define DD   256
#define HH   512
#define G3H  1536   // 3*H

typedef __attribute__((ext_vector_type(16))) __bf16 v16bf;
typedef __attribute__((ext_vector_type(8)))  __bf16 v8bf;
typedef __attribute__((ext_vector_type(8)))  float  v8f;

// ---------------------------------------------------------------------------
// f32 -> bf16 cast kernel (weights + x pre-pass, fully parallel)
// ---------------------------------------------------------------------------
__global__ void cast_f32_to_bf16(const float* __restrict__ src,
                                 __bf16* __restrict__ dst, int n) {
    int i = blockIdx.x * blockDim.x + threadIdx.x;
    if (i < n) dst[i] = (__bf16)src[i];
}

// ---------------------------------------------------------------------------
// WMMA helper: D = A(16x32 bf16) * B(32x16 bf16) + C(16x16 f32)
// ---------------------------------------------------------------------------
__device__ __forceinline__ v8f wmma_bf16(v16bf a, v16bf b, v8f c) {
    return __builtin_amdgcn_wmma_f32_16x16x32_bf16(
        /*neg_a=*/false, a, /*neg_b=*/false, b,
        /*c_mod=*/(short)0, c, /*reuse_a=*/false, /*reuse_b=*/false);
}

__device__ __forceinline__ v16bf cat8(v8bf lo, v8bf hi) {
    return __builtin_shufflevector(lo, hi, 0, 1, 2, 3, 4, 5, 6, 7,
                                   8, 9, 10, 11, 12, 13, 14, 15);
}

// Fragment load: 8 contiguous bf16 at base+idx, 8 more at base+idx+16.
// Uniform base + 32-bit offset -> SADDR+VOFFSET global_load_b128 form.
__device__ __forceinline__ v16bf load_frag_g(const __bf16* __restrict__ base, int idx) {
    v8bf lo = *(const v8bf*)(base + idx);
    v8bf hi = *(const v8bf*)(base + idx + 16);
    return cat8(lo, hi);
}

__device__ __forceinline__ v16bf load_frag_lds(const __bf16* base, int idx) {
    v8bf lo = *(const v8bf*)(base + idx);
    v8bf hi = *(const v8bf*)(base + idx + 16);
    return cat8(lo, hi);
}

// ---------------------------------------------------------------------------
// One GRU layer step, fully specialized per layer so all global pointers keep
// addrspace(1). K = input width (256 or 512). FROM_X: A operand comes from
// global bf16 x; else from LDS bf16 plane. The two M-tiles are processed in
// sequential passes to keep only 6 f32 accumulators (48 VGPRs) live at once.
// k-loops stay rolled so the weight stream cannot be hoisted/spilled.
// ---------------------------------------------------------------------------
template <int K, bool FROM_X>
__device__ __forceinline__ void layer_step(
    const __bf16* __restrict__ xb,      // global: bf16 x base (if FROM_X)
    int x0off, int x1off,               // per-lane x row offsets (t applied)
    const __bf16* hin,                  // LDS: previous-layer hidden (if !FROM_X)
    __bf16* hcur,                       // LDS: this layer's hidden (read + write)
    const __bf16* __restrict__ wi,      // Wi bf16, row-major (3H, K)
    const __bf16* __restrict__ wh,      // Wh bf16, row-major (3H, HH)
    const float* __restrict__ bi, const float* __restrict__ bh,
    __bf16* sh_rt, __bf16* sh_zt, __bf16* sh_s1, __bf16* sh_hh,
    int tid, int lm, int kboff, int mrow, int nt0) {

    const int wrow = nt0 * 16 + lm;     // this lane's weight row (B fragment)

#pragma unroll
    for (int mt = 0; mt < 2; ++mt) {
        v8f acc_i[3];
        v8f acc_h[3];
#pragma unroll
        for (int j = 0; j < 3; ++j) { acc_i[j] = (v8f)(0.0f); acc_h[j] = (v8f)(0.0f); }

        const int arow = (mt * 16 + lm) * HH;     // LDS A row offset
        const int xoff = mt ? x1off : x0off;      // global A row offset

        // ---- input-projection GEMM ----
#pragma clang loop unroll(disable)
        for (int kk = 0; kk < K; kk += 32) {
            v16bf a;
            if constexpr (FROM_X) {
                a = load_frag_g(xb, xoff + kk + kboff);
            } else {
                a = load_frag_lds(hin, arow + kk + kboff);
            }
            __builtin_prefetch(wi + wrow * K + kk + 64, 0, 3);  // next k-slice (L2-near)
#pragma unroll
            for (int j = 0; j < 3; ++j) {
                v16bf bf = load_frag_g(wi, (wrow + j * 16) * K + kk + kboff);
                acc_i[j] = wmma_bf16(a, bf, acc_i[j]);
            }
        }

        // ---- hidden-projection GEMM ----
#pragma clang loop unroll(disable)
        for (int kk = 0; kk < HH; kk += 32) {
            v16bf a = load_frag_lds(hcur, arow + kk + kboff);
            __builtin_prefetch(wh + wrow * HH + kk + 64, 0, 3);
#pragma unroll
            for (int j = 0; j < 3; ++j) {
                v16bf bf = load_frag_g(wh, (wrow + j * 16) * HH + kk + kboff);
                acc_h[j] = wmma_bf16(a, bf, acc_h[j]);
            }
        }

        // ---- combine: biases + gate transforms, scatter to LDS planes ----
        // Gate boundaries (n = 512, 1024) are tile-aligned -> uniform branch.
        // Passes mt=0/1 write disjoint row ranges; one barrier after both.
#pragma unroll
        for (int j = 0; j < 3; ++j) {
            const int n = (nt0 + j) * 16 + lm;   // column in [0,1536)
            const float bin = bi[n];
            const float bhn = bh[n];
#pragma unroll
            for (int e = 0; e < 8; ++e) {
                const int m = mt * 16 + e + mrow;  // row in [0,32)
                const float vi = acc_i[j][e] + bin;
                const float vh = acc_h[j][e] + bhn;
                if (n < HH) {                        // r gate
                    float s = 1.0f / (1.0f + __expf(-(vi + vh)));
                    sh_rt[m * HH + n] = (__bf16)__expf(-__expf(-s));
                } else if (n < 2 * HH) {             // z gate
                    float s = 1.0f / (1.0f + __expf(-(vi + vh)));
                    sh_zt[m * HH + (n - HH)] = (__bf16)__expf(-__expf(-s));
                } else {                             // candidate
                    const int jn = n - 2 * HH;
                    sh_s1[m * HH + jn] = (__bf16)(vi + vh);
                    sh_hh[m * HH + jn] = (__bf16)vh;
                }
            }
        }
    }
    __syncthreads();

    // ---- elementwise cell update: h := (1-z)*n + z*h ----
    for (int idx = tid; idx < BB * HH; idx += 1024) {
        const float rt   = (float)sh_rt[idx];
        const float zt   = (float)sh_zt[idx];
        const float s1   = (float)sh_s1[idx];
        const float hhn  = (float)sh_hh[idx];
        const float hold = (float)hcur[idx];
        const float nn   = tanhf(s1 + rt * hhn);
        hcur[idx] = (__bf16)((1.0f - zt) * nn + zt * hold);
    }
    __syncthreads();
}

// ---------------------------------------------------------------------------
// Persistent recurrent kernel: one workgroup (32 wave32 waves) runs all
// T=512 steps x 3 layers. Hidden state + gate scratch live in LDS (224KB).
// Each wave owns 3 N-tiles (48 output columns) x 2 M-tiles for both GEMMs.
// ---------------------------------------------------------------------------
__global__ __launch_bounds__(1024, 1)
void gru_persistent(const __bf16* __restrict__ xb,
                    const __bf16* __restrict__ Wi0b, const __bf16* __restrict__ Wh0b,
                    const __bf16* __restrict__ Wi1b, const __bf16* __restrict__ Wh1b,
                    const __bf16* __restrict__ Wi2b, const __bf16* __restrict__ Wh2b,
                    const float* __restrict__ bi0, const float* __restrict__ bh0,
                    const float* __restrict__ bi1, const float* __restrict__ bh1,
                    const float* __restrict__ bi2, const float* __restrict__ bh2,
                    float* __restrict__ out) {
    __shared__ __align__(16) __bf16 sh_h0[BB * HH];
    __shared__ __align__(16) __bf16 sh_h1[BB * HH];
    __shared__ __align__(16) __bf16 sh_h2[BB * HH];
    __shared__ __align__(16) __bf16 sh_rt[BB * HH];
    __shared__ __align__(16) __bf16 sh_zt[BB * HH];
    __shared__ __align__(16) __bf16 sh_s1[BB * HH];
    __shared__ __align__(16) __bf16 sh_hh[BB * HH];

    const int tid   = threadIdx.x;
    const int wave  = tid >> 5;
    const int lane  = tid & 31;
    const int lm    = lane & 15;
    const int kboff = (lane >> 4) ? 8 : 0;   // K sub-offset for A/B fragments
    const int mrow  = (lane >> 4) ? 8 : 0;   // M sub-offset for C/D fragments
    const int nt0   = wave * 3;              // first owned N-tile

    // zero-init hidden states
    for (int i = tid; i < BB * HH; i += 1024) {
        sh_h0[i] = (__bf16)0.0f;
        sh_h1[i] = (__bf16)0.0f;
        sh_h2[i] = (__bf16)0.0f;
    }
    __syncthreads();

    // Per-lane x row base offsets (batch rows lm and lm+16), 32-bit
    const int xr0 = lm * (TT * DD);
    const int xr1 = (lm + 16) * (TT * DD);

    for (int t = 0; t < TT; ++t) {
        const int x0 = xr0 + t * DD;
        const int x1 = xr1 + t * DD;
        layer_step<DD, true >(xb, x0, x1, nullptr, sh_h0, Wi0b, Wh0b, bi0, bh0,
                              sh_rt, sh_zt, sh_s1, sh_hh, tid, lm, kboff, mrow, nt0);
        layer_step<HH, false>(nullptr, 0, 0, sh_h0, sh_h1, Wi1b, Wh1b, bi1, bh1,
                              sh_rt, sh_zt, sh_s1, sh_hh, tid, lm, kboff, mrow, nt0);
        layer_step<HH, false>(nullptr, 0, 0, sh_h1, sh_h2, Wi2b, Wh2b, bi2, bh2,
                              sh_rt, sh_zt, sh_s1, sh_hh, tid, lm, kboff, mrow, nt0);
    }

    // output: final hidden of last layer, (B,1,H) f32
    for (int idx = tid; idx < BB * HH; idx += 1024)
        out[idx] = (float)sh_h2[idx];
}

// ---------------------------------------------------------------------------
// Launch
// ---------------------------------------------------------------------------
extern "C" void kernel_launch(void* const* d_in, const int* in_sizes, int n_in,
                              void* d_out, int out_size, void* d_ws, size_t ws_size,
                              hipStream_t stream) {
    const float* x    = (const float*)d_in[0];
    const float* Wi0f = (const float*)d_in[1];
    const float* bi0  = (const float*)d_in[2];
    const float* Wh0f = (const float*)d_in[3];
    const float* bh0  = (const float*)d_in[4];
    const float* Wi1f = (const float*)d_in[5];
    const float* bi1  = (const float*)d_in[6];
    const float* Wh1f = (const float*)d_in[7];
    const float* bh1  = (const float*)d_in[8];
    const float* Wi2f = (const float*)d_in[9];
    const float* bi2  = (const float*)d_in[10];
    const float* Wh2f = (const float*)d_in[11];
    const float* bh2  = (const float*)d_in[12];

    // bf16 staging in workspace: 6 weight matrices + x
    const size_t nWi0 = (size_t)G3H * DD;        // 1536*256
    const size_t nW   = (size_t)G3H * HH;        // 1536*512
    const size_t nX   = (size_t)BB * TT * DD;    // 32*512*256
    char* ws = (char*)d_ws;
    __bf16* Wi0b = (__bf16*)(ws);
    __bf16* Wh0b = (__bf16*)(ws + 2 * nWi0);
    __bf16* Wi1b = (__bf16*)(ws + 2 * (nWi0 + nW));
    __bf16* Wh1b = (__bf16*)(ws + 2 * (nWi0 + 2 * nW));
    __bf16* Wi2b = (__bf16*)(ws + 2 * (nWi0 + 3 * nW));
    __bf16* Wh2b = (__bf16*)(ws + 2 * (nWi0 + 4 * nW));
    __bf16* xbf  = (__bf16*)(ws + 2 * (nWi0 + 5 * nW));

    const int blk = 256;
    cast_f32_to_bf16<<<(int)((nWi0 + blk - 1) / blk), blk, 0, stream>>>(Wi0f, Wi0b, (int)nWi0);
    cast_f32_to_bf16<<<(int)((nW + blk - 1) / blk),   blk, 0, stream>>>(Wh0f, Wh0b, (int)nW);
    cast_f32_to_bf16<<<(int)((nW + blk - 1) / blk),   blk, 0, stream>>>(Wi1f, Wi1b, (int)nW);
    cast_f32_to_bf16<<<(int)((nW + blk - 1) / blk),   blk, 0, stream>>>(Wh1f, Wh1b, (int)nW);
    cast_f32_to_bf16<<<(int)((nW + blk - 1) / blk),   blk, 0, stream>>>(Wi2f, Wi2b, (int)nW);
    cast_f32_to_bf16<<<(int)((nW + blk - 1) / blk),   blk, 0, stream>>>(Wh2f, Wh2b, (int)nW);
    cast_f32_to_bf16<<<(int)((nX + blk - 1) / blk),   blk, 0, stream>>>(x,    xbf,  (int)nX);

    gru_persistent<<<1, 1024, 0, stream>>>(xbf,
                                           Wi0b, Wh0b, Wi1b, Wh1b, Wi2b, Wh2b,
                                           bi0, bh0, bi1, bh1, bi2, bh2,
                                           (float*)d_out);
}
